// vgae_decoder_14800457302103
// MI455X (gfx1250) — matrix operand order repaired and checked by reading, not compiled, not run on hardware
//
#include <hip/hip_runtime.h>

typedef __attribute__((ext_vector_type(16))) __bf16        v16bf;
typedef __attribute__((ext_vector_type(8)))  float         v8f;
typedef __attribute__((ext_vector_type(4)))  float         f32x4;
typedef __attribute__((ext_vector_type(4)))  unsigned int  u32x4;

#define HID    128
#define TILE_M 128
#define ROWB   272   // LDS bytes per H row: 4 ksteps * 64B + 16B pad (bank-conflict relief)

__device__ __forceinline__ unsigned int pack_bf16(float a, float b) {
  // round-to-nearest-even fp32 -> bf16, pack pair (low = even-K element)
  unsigned int ua = __float_as_uint(a);
  unsigned int ub = __float_as_uint(b);
  ua = (ua + 0x7FFFu + ((ua >> 16) & 1u)) >> 16;
  ub = (ub + 0x7FFFu + ((ub >> 16) & 1u)) >> 16;
  return (ub << 16) | (ua & 0xFFFFu);
}

union Frag {
  v16bf        v;
  u32x4        q[2];
  unsigned int w[8];
};

__global__ __launch_bounds__(256)
void vgae_decoder_kernel(const float* __restrict__ x,  const float* __restrict__ W1,
                         const float* __restrict__ b1, const float* __restrict__ W2,
                         const float* __restrict__ b2, const int* __restrict__ ei,
                         const int* __restrict__ ein,  float* __restrict__ out,
                         int E, int Mtotal, int tiles)
{
  __shared__ __align__(16) unsigned char Hraw[TILE_M * ROWB];
  __shared__ float rowacc[TILE_M];

  const int t    = threadIdx.x;
  const int wv   = t >> 5;          // wave id == n-tile (0..7)
  const int lane = t & 31;
  const int nl   = lane & 15;
  const int half = lane >> 4;
  const int ncol = (wv << 4) + nl;  // this lane's output column in [0,128)

  // ---- per-block preload: W1 as bf16 B-fragments (registers), b1/W2/b2 ----
  // B k order used here (contiguous, matching A): lane holds column n = L%16; lanes 0-15 K=0..15,
  // lanes 16-31 K=16..31; dword d = K pair {2d, 2d+1} (+16*half, +32*kstep)
  Frag B[4];
  #pragma unroll
  for (int ks = 0; ks < 4; ++ks) {
    #pragma unroll
    for (int d = 0; d < 8; ++d) {
      const int k0 = (ks << 5) + (half << 4) + (d << 1);
      const float f0 = W1[k0 * HID + ncol];
      const float f1 = W1[(k0 + 1) * HID + ncol];
      B[ks].w[d] = pack_bf16(f0, f1);
    }
  }
  const float b1v = b1[ncol];
  const float w2v = W2[ncol];
  const float b2s = b2[0];

  // slot -> K permutation inside one 32-wide kstep block: stores land so that
  // an A-fragment (half h) is two contiguous 16B chunks => 2x ds_load_b128
  const int tbl[16] = {0,2,4,6,8,10,12,14, 16,18,20,22,24,26,28,30};

  for (int tile = blockIdx.x; tile < tiles; tile += gridDim.x) {
    if (t < TILE_M) rowacc[t] = 0.0f;

    // ---- phase 1: gather endpoints, hadamard, relu, bf16 -> LDS ----
    {
      const int r  = t >> 1;   // edge row within tile
      const int ch = t & 1;    // which 64-column half this thread fills
      int g = tile * TILE_M + r;
      if (g >= Mtotal) g = 0;
      int s, d2;
      if (g < E) { s = ei[g];      d2 = ei[E + g]; }
      else       { s = ein[g - E]; d2 = ein[g];    }
      const f32x4* xs = reinterpret_cast<const f32x4*>(x + (long)s  * HID);
      const f32x4* xd = reinterpret_cast<const f32x4*>(x + (long)d2 * HID);
      #pragma unroll
      for (int ks2 = 0; ks2 < 2; ++ks2) {
        const int ks = (ch << 1) + ks2;
        float p[32];
        #pragma unroll
        for (int i = 0; i < 8; ++i) {
          const f32x4 a = xs[(ks << 3) + i];
          const f32x4 c = xd[(ks << 3) + i];
          p[4*i+0] = fmaxf(a.x * c.x, 0.0f);
          p[4*i+1] = fmaxf(a.y * c.y, 0.0f);
          p[4*i+2] = fmaxf(a.z * c.z, 0.0f);
          p[4*i+3] = fmaxf(a.w * c.w, 0.0f);
        }
        union { unsigned int w[16]; u32x4 q[4]; } blk;
        #pragma unroll
        for (int j = 0; j < 16; ++j) {
          const int k = tbl[j];
          blk.w[j] = pack_bf16(p[k], p[k + 1]);
        }
        u32x4* dst = reinterpret_cast<u32x4*>(&Hraw[r * ROWB + (ks << 6)]);
        dst[0] = blk.q[0]; dst[1] = blk.q[1]; dst[2] = blk.q[2]; dst[3] = blk.q[3];
      }
    }
    __syncthreads();

    // ---- phase 2: Z = relu-H @ W1, bf16 WMMA, wave wv owns 128x16 slab ----
    v8f acc[8];
    const v8f vz = {0.f,0.f,0.f,0.f,0.f,0.f,0.f,0.f};
    #pragma unroll
    for (int m = 0; m < 8; ++m) acc[m] = vz;

    #pragma unroll
    for (int ks = 0; ks < 4; ++ks) {
      #pragma unroll
      for (int m = 0; m < 8; ++m) {
        Frag A;
        const u32x4* ap = reinterpret_cast<const u32x4*>(
            &Hraw[((m << 4) + nl) * ROWB + (ks << 6) + (half << 5)]);
        A.q[0] = ap[0];
        A.q[1] = ap[1];
        acc[m] = __builtin_amdgcn_wmma_f32_16x16x32_bf16(
            false, A.v, false, B[ks].v, (short)0, acc[m], false, false);
      }
    }

    // ---- phase 3: +b1, relu, dot W2; reduce lanes (shfl) + waves (ds_add) ----
    #pragma unroll
    for (int m = 0; m < 8; ++m) {
      #pragma unroll
      for (int v = 0; v < 8; ++v) {
        const float z  = fmaxf(acc[m][v] + b1v, 0.0f);
        float pz = z * w2v;
        pz += __shfl_xor(pz, 1);
        pz += __shfl_xor(pz, 2);
        pz += __shfl_xor(pz, 4);
        pz += __shfl_xor(pz, 8);
        if (nl == 0) atomicAdd(&rowacc[(m << 4) + (half << 3) + v], pz);
      }
    }
    __syncthreads();

    // ---- phase 4: sigmoid + store ----
    if (t < TILE_M) {
      const int g = tile * TILE_M + t;
      if (g < Mtotal) {
        const float zz = rowacc[t] + b2s;
        out[g] = 1.0f / (1.0f + __expf(-zz));
      }
    }
    __syncthreads();
  }
}

extern "C" void kernel_launch(void* const* d_in, const int* in_sizes, int n_in,
                              void* d_out, int out_size, void* d_ws, size_t ws_size,
                              hipStream_t stream) {
  (void)n_in; (void)out_size; (void)d_ws; (void)ws_size;
  const float* x  = (const float*)d_in[0];
  const float* W1 = (const float*)d_in[1];
  const float* b1 = (const float*)d_in[2];
  const float* W2 = (const float*)d_in[3];
  const float* b2 = (const float*)d_in[4];
  const int*   ei  = (const int*)d_in[5];
  const int*   ein = (const int*)d_in[6];
  float* out = (float*)d_out;

  const int Mtotal = in_sizes[5];       // edge_index is [2,E] flat -> 2E rows total (pos+neg)
  const int E      = Mtotal / 2;
  const int tiles  = (Mtotal + TILE_M - 1) / TILE_M;
  const int grid   = tiles < 1280 ? tiles : 1280;

  vgae_decoder_kernel<<<dim3(grid), dim3(256), 0, stream>>>(
      x, W1, b1, W2, b2, ei, ein, out, E, Mtotal, tiles);
}
